// MultiheadSelfAttention_68951404970432
// MI455X (gfx1250) — matrix operand order repaired and checked by reading, not compiled, not run on hardware
//
#include <hip/hip_runtime.h>
#include <hip/hip_bf16.h>

// ---------------------------------------------------------------------------
// Types and WMMA helpers (gfx1250 / CDNA5, wave32)
// ---------------------------------------------------------------------------
typedef __attribute__((ext_vector_type(16))) __bf16 v16bf;
typedef __attribute__((ext_vector_type(8)))  float  v8f;

union FragAB { int i[8]; v16bf v; };

// A-matrix fragment, 16(M) x 32(K) bf16, row-major source with leading dim `ld`.
// ISA layout: lanes 0-15 = rows 0-15; v0..v3: K = kh*8 + {0..7}; v4..v7: K = 16 + kh*8 + {0..7};
// kh = lane>>4. Two bf16 per VGPR, even K in [15:0].
__device__ __forceinline__ v16bf load_a16x32(const __bf16* p, int ld) {
  const int lane = threadIdx.x & 31;
  const int m = lane & 15, kh = lane >> 4;
  const __bf16* row = p + (size_t)m * ld + kh * 8;
  FragAB f;
#pragma unroll
  for (int r = 0; r < 4; ++r) {
    f.i[r]     = *(const int*)(row + 2 * r);
    f.i[r + 4] = *(const int*)(row + 16 + 2 * r);
  }
  return f.v;
}

// B-matrix fragment, 32(K) x 16(N) bf16 where B[k][n] = p[n*ld + k]
// (i.e. the K dimension is contiguous per n-row — weights / K-rows / V^T rows).
// Layout: lane = n (0-15) | n (16-31); v0..v3: K = kh*8 + {0..7}; v4..v7: K = 16 + kh*8 + {0..7}.
__device__ __forceinline__ v16bf load_b32x16(const __bf16* p, int ld) {
  const int lane = threadIdx.x & 31;
  const int n = lane & 15, kh = lane >> 4;
  const __bf16* row = p + (size_t)n * ld + kh * 8;
  FragAB f;
#pragma unroll
  for (int r = 0; r < 4; ++r) {
    f.i[r]     = *(const int*)(row + 2 * r);
    f.i[r + 4] = *(const int*)(row + 16 + 2 * r);
  }
  return f.v;
}

__device__ __forceinline__ v8f wmma_bf16(v16bf a, v16bf b, v8f c) {
  // D = A(16x32) * B(32x16) + C, f32 accumulate
  return __builtin_amdgcn_wmma_f32_16x16x32_bf16(false, a, false, b, (short)0, c,
                                                 false, false);
}

// ---------------------------------------------------------------------------
// Problem constants
// ---------------------------------------------------------------------------
#define NB   4
#define SEQ  2048
#define DIM  1024
#define NH   16
#define DK   64

// ---------------------------------------------------------------------------
// Kernel 1: fp32 -> bf16 conversion
// ---------------------------------------------------------------------------
__global__ void cvt_f32_to_bf16(const float* __restrict__ src,
                                __bf16* __restrict__ dst, int n) {
  int i = blockIdx.x * blockDim.x + threadIdx.x;
  const int stride = gridDim.x * blockDim.x;
  for (; i < n; i += stride) dst[i] = (__bf16)src[i];
}

// ---------------------------------------------------------------------------
// Kernel 2: fused QKV projection (Y = X * W^T) with RoPE epilogue.
// Each wave: 16 rows x 64 cols (= one head's dk). blockIdx.z selects Q/K/V.
// Q,K stored [b,h,s,dk]; V stored transposed [b,h,dk,s].
// ---------------------------------------------------------------------------
__global__ void __launch_bounds__(128)
qkv_proj_rope_kernel(const __bf16* __restrict__ xb,
                     const __bf16* __restrict__ Wqb,
                     const __bf16* __restrict__ Wkb,
                     const __bf16* __restrict__ Wvb,
                     __bf16* __restrict__ Qb,
                     __bf16* __restrict__ Kb,
                     __bf16* __restrict__ Vtb) {
  const int wave  = threadIdx.x >> 5;
  const int lane  = threadIdx.x & 31;
  const int mtile = blockIdx.x * 4 + wave;          // 0..511  (B*S/16)
  const int h     = blockIdx.y;                     // head
  const int which = blockIdx.z;                     // 0=Q 1=K 2=V

  const __bf16* W  = (which == 0) ? Wqb : (which == 1) ? Wkb : Wvb;
  const __bf16* A  = xb + (size_t)mtile * 16 * DIM; // 16 rows of x
  const __bf16* Bw = W + (size_t)h * DK * DIM;      // rows e = h*64 .. h*64+63

  v8f acc[4] = {};
  for (int k0 = 0; k0 < DIM; k0 += 32) {
    v16bf a = load_a16x32(A + k0, DIM);
#pragma unroll
    for (int t = 0; t < 4; ++t) {
      v16bf b = load_b32x16(Bw + (size_t)t * 16 * DIM + k0, DIM);
      acc[t] = wmma_bf16(a, b, acc[t]);
    }
  }

  const int n = lane & 15, mb = (lane >> 4) * 8;
#pragma unroll
  for (int t = 0; t < 4; ++t) {
#pragma unroll
    for (int r = 0; r < 8; ++r) {
      const int row = mtile * 16 + mb + r;          // global row in [B*S)
      const int b   = row >> 11;                    // / SEQ
      const int s   = row & (SEQ - 1);
      const int d   = t * 16 + n;                   // 0..63 within head
      float val = acc[t][r];
      if (which == 2) {
        // V transposed: Vt[b][h][d][s]
        Vtb[((size_t)(b * NH + h) * DK + d) * SEQ + s] = (__bf16)val;
      } else {
        // RoPE: pairs (2i, 2i+1) along d; partner element lives in lane^1.
        const float inv = __expf((float)(d >> 1) * -0.2878231366f); // ln(1e4)/32
        const float ang = (float)s * inv;
        const float cs = __cosf(ang), sn = __sinf(ang);
        const float partner = __shfl_xor(val, 1, 32);
        // d even: x1*c - x2*s ; d odd: x1*s + x2*c
        const float out = val * cs + ((n & 1) ? partner * sn : -partner * sn);
        __bf16* dst = (which == 0) ? Qb : Kb;
        dst[((size_t)(b * NH + h) * SEQ + s) * DK + d] = (__bf16)out;
      }
    }
  }
}

// ---------------------------------------------------------------------------
// Kernel 3: causal flash attention. One wave per 16-query tile per (b,h).
// Streams 32-key chunks: 4 WMMA scores + fp32 online softmax + 4 WMMA for P*V.
// ---------------------------------------------------------------------------
__global__ void __launch_bounds__(128)
flash_attn_kernel(const __bf16* __restrict__ Qb,
                  const __bf16* __restrict__ Kb,
                  const __bf16* __restrict__ Vtb,
                  __bf16* __restrict__ ctx) {
  constexpr int PLD = 40;                           // padded LDS row stride
  __shared__ __bf16 pst[4][16 * PLD];

  const int wave = threadIdx.x >> 5;
  const int lane = threadIdx.x & 31;
  const int qt = blockIdx.x * 4 + wave;             // 0..127 query tiles
  const int h  = blockIdx.y;
  const int b  = blockIdx.z;

  const __bf16* Qh = Qb  + (size_t)(b * NH + h) * SEQ * DK;
  const __bf16* Kh = Kb  + (size_t)(b * NH + h) * SEQ * DK;
  const __bf16* Vh = Vtb + (size_t)(b * NH + h) * DK * SEQ;

  // Q tile held in registers for the whole kv loop (d=0..31, d=32..63)
  const v16bf qa0 = load_a16x32(Qh + (size_t)qt * 16 * DK, DK);
  const v16bf qa1 = load_a16x32(Qh + (size_t)qt * 16 * DK + 32, DK);

  const int n = lane & 15, mb = (lane >> 4) * 8;
  float mrow[8], lrow[8];
  v8f o[4] = {};
#pragma unroll
  for (int r = 0; r < 8; ++r) { mrow[r] = -3.0e38f; lrow[r] = 0.0f; }

  const int nch = (qt * 16 + 15) / 32 + 1;          // causal chunk bound
  for (int jc = 0; jc < nch; ++jc) {
    const int kbase = jc * 32;
    __builtin_prefetch(Kh + (size_t)(kbase + 32) * DK, 0, 1);
    __builtin_prefetch(Vh + (size_t)(kbase + 32), 0, 1);

    // ---- scores S = Q * K^T (two 16x16 tiles, K-dim 64 = 2 x 32) ----
    v8f sc[2] = {};
#pragma unroll
    for (int st = 0; st < 2; ++st) {
      const __bf16* Kt = Kh + (size_t)(kbase + st * 16) * DK;
      v16bf bk0 = load_b32x16(Kt, DK);
      v16bf bk1 = load_b32x16(Kt + 32, DK);
      sc[st] = wmma_bf16(qa0, bk0, sc[st]);
      sc[st] = wmma_bf16(qa1, bk1, sc[st]);
    }

    // ---- scale + causal mask ----
#pragma unroll
    for (int st = 0; st < 2; ++st)
#pragma unroll
      for (int r = 0; r < 8; ++r) {
        const int kj = kbase + st * 16 + n;
        const int qr = qt * 16 + mb + r;
        const float v = sc[st][r] * 0.125f;         // 1/sqrt(64)
        sc[st][r] = (kj > qr) ? -3.0e38f : v;
      }

    // ---- online softmax (row reductions across the 16-lane half) ----
#pragma unroll
    for (int r = 0; r < 8; ++r) {
      float rmax = fmaxf(sc[0][r], sc[1][r]);
#pragma unroll
      for (int msk = 1; msk < 16; msk <<= 1)
        rmax = fmaxf(rmax, __shfl_xor(rmax, msk, 32));
      const float mn   = fmaxf(mrow[r], rmax);
      const float corr = __expf(mrow[r] - mn);
      mrow[r] = mn;
      const float p0 = __expf(sc[0][r] - mn);
      const float p1 = __expf(sc[1][r] - mn);
      sc[0][r] = p0; sc[1][r] = p1;
      float rs = p0 + p1;
#pragma unroll
      for (int msk = 1; msk < 16; msk <<= 1)
        rs += __shfl_xor(rs, msk, 32);
      lrow[r] = lrow[r] * corr + rs;
#pragma unroll
      for (int t = 0; t < 4; ++t) o[t][r] *= corr;
    }

    // ---- restage P (C layout) through LDS into A-fragment layout ----
    __bf16* ps = &pst[wave][0];
#pragma unroll
    for (int st = 0; st < 2; ++st)
#pragma unroll
      for (int r = 0; r < 8; ++r)
        ps[(mb + r) * PLD + st * 16 + n] = (__bf16)sc[st][r];
    // same-wave LDS RAW: hardware DScnt keeps this in order (compiler waits)
    const v16bf pa = load_a16x32(ps, PLD);

    // ---- O += P * V (V^T rows are contiguous in k) ----
#pragma unroll
    for (int t = 0; t < 4; ++t) {
      v16bf bv = load_b32x16(Vh + (size_t)t * 16 * SEQ + kbase, SEQ);
      o[t] = wmma_bf16(pa, bv, o[t]);
    }
  }

  // ---- normalize + store context [b, s, e] ----
#pragma unroll
  for (int t = 0; t < 4; ++t)
#pragma unroll
    for (int r = 0; r < 8; ++r) {
      const int s = qt * 16 + mb + r;
      const int e = h * DK + t * 16 + n;
      const float val = o[t][r] / lrow[r];
      ctx[((size_t)b * SEQ + s) * DIM + e] = (__bf16)val;
    }
}

// ---------------------------------------------------------------------------
// Kernel 4: output projection out = ctx * Wo^T, fp32 result to d_out.
// ---------------------------------------------------------------------------
__global__ void __launch_bounds__(128)
out_proj_kernel(const __bf16* __restrict__ ctx,
                const __bf16* __restrict__ Wob,
                float* __restrict__ out) {
  const int wave  = threadIdx.x >> 5;
  const int lane  = threadIdx.x & 31;
  const int mtile = blockIdx.x * 4 + wave;          // 0..511
  const int nt    = blockIdx.y;                     // 64-col tile, 0..15

  const __bf16* A  = ctx + (size_t)mtile * 16 * DIM;
  const __bf16* Bw = Wob + (size_t)nt * 64 * DIM;

  v8f acc[4] = {};
  for (int k0 = 0; k0 < DIM; k0 += 32) {
    v16bf a = load_a16x32(A + k0, DIM);
#pragma unroll
    for (int t = 0; t < 4; ++t) {
      v16bf b = load_b32x16(Bw + (size_t)t * 16 * DIM + k0, DIM);
      acc[t] = wmma_bf16(a, b, acc[t]);
    }
  }

  const int n = lane & 15, mb = (lane >> 4) * 8;
#pragma unroll
  for (int t = 0; t < 4; ++t)
#pragma unroll
    for (int r = 0; r < 8; ++r)
      out[(size_t)(mtile * 16 + mb + r) * DIM + nt * 64 + t * 16 + n] = acc[t][r];
}

// ---------------------------------------------------------------------------
// Host launcher
// ---------------------------------------------------------------------------
extern "C" void kernel_launch(void* const* d_in, const int* in_sizes, int n_in,
                              void* d_out, int out_size, void* d_ws, size_t ws_size,
                              hipStream_t stream) {
  (void)n_in; (void)out_size; (void)ws_size;
  const float* x  = (const float*)d_in[0];
  const float* Wq = (const float*)d_in[1];
  const float* Wk = (const float*)d_in[2];
  const float* Wv = (const float*)d_in[3];
  const float* Wo = (const float*)d_in[4];
  float* out = (float*)d_out;

  const size_t BSD = (size_t)NB * SEQ * DIM;   // 8388608
  const size_t DD  = (size_t)DIM * DIM;        // 1048576

  char* wsp = (char*)d_ws;
  size_t off = 0;
  auto take = [&](size_t bytes) -> void* {
    void* p = wsp + off;
    off += (bytes + 255) & ~(size_t)255;
    return p;
  };
  __bf16* xb  = (__bf16*)take(BSD * 2);
  __bf16* Wqb = (__bf16*)take(DD * 2);
  __bf16* Wkb = (__bf16*)take(DD * 2);
  __bf16* Wvb = (__bf16*)take(DD * 2);
  __bf16* Wob = (__bf16*)take(DD * 2);
  __bf16* Qb  = (__bf16*)take(BSD * 2);        // [b,h,s,dk]
  __bf16* Kb  = (__bf16*)take(BSD * 2);        // [b,h,s,dk]
  __bf16* Vtb = (__bf16*)take(BSD * 2);        // [b,h,dk,s]
  __bf16* ctx = (__bf16*)take(BSD * 2);        // [b,s,e]

  cvt_f32_to_bf16<<<dim3(2048), 256, 0, stream>>>(x,  xb,  (int)BSD);
  cvt_f32_to_bf16<<<dim3(512),  256, 0, stream>>>(Wq, Wqb, (int)DD);
  cvt_f32_to_bf16<<<dim3(512),  256, 0, stream>>>(Wk, Wkb, (int)DD);
  cvt_f32_to_bf16<<<dim3(512),  256, 0, stream>>>(Wv, Wvb, (int)DD);
  cvt_f32_to_bf16<<<dim3(512),  256, 0, stream>>>(Wo, Wob, (int)DD);

  // 512 M-tiles / 4 waves per block, 16 heads, {Q,K,V}
  qkv_proj_rope_kernel<<<dim3(128, 16, 3), 128, 0, stream>>>(
      xb, Wqb, Wkb, Wvb, Qb, Kb, Vtb);

  // 128 query tiles / 4 waves per block, 16 heads, 4 batches
  flash_attn_kernel<<<dim3(32, 16, 4), 128, 0, stream>>>(Qb, Kb, Vtb, ctx);

  // 512 M-tiles / 4 waves per block, 16 column tiles of 64
  out_proj_kernel<<<dim3(128, 16), 128, 0, stream>>>(ctx, Wob, out);
}
